// MoELayer_24275155157558
// MI455X (gfx1250) — matrix operand order repaired and checked by reading, not compiled, not run on hardware
//
#include <hip/hip_runtime.h>
#include <hip/hip_bf16.h>
#include <math.h>

// ---------------- problem constants (from reference) ----------------
#define TOK_N   2048      // B*S tokens
#define DIM_H   1024
#define NEXP    8
#define DIM_I   512
#define DIM_IS  2048
#define TOPK    4
#define MAXR    2048      // max tokens routed to one expert (<= N)

// ---------------- tiling ----------------
#define TILE_M  64
#define TILE_N  128
#define TILE_K  64
#define LDX     88        // LDS leading dim (bf16 elems): 176B rows -> 16B aligned, 44-bank stride
#define LDB     88

typedef __attribute__((ext_vector_type(16))) __bf16 v16bf;
typedef __attribute__((ext_vector_type(8)))  float  v8f;
typedef __attribute__((ext_vector_type(4)))  int    v4i;

// ---- gfx1250 async global->LDS copy (ASYNCcnt path), guarded so absence degrades gracefully ----
// Probe round 2 revealed the builtin signature: (int4 addrspace(1)*, int4 addrspace(3)*, imm, imm)
#if __has_builtin(__builtin_amdgcn_global_load_async_to_lds_b128)
#define HAVE_ASYNC_LDS 1
#define GLB_V4I(p) ((__attribute__((address_space(1))) v4i*)(p))
#define LDS_V4I(p) ((__attribute__((address_space(3))) v4i*)(p))
#endif

__device__ __forceinline__ void wait_async0() {
#if __has_builtin(__builtin_amdgcn_s_wait_asynccnt)
  __builtin_amdgcn_s_wait_asynccnt(0);
#else
  asm volatile("s_wait_asynccnt 0x0" ::: "memory");
#endif
}

__device__ __forceinline__ v8f zero8() {
  v8f z;
#pragma unroll
  for (int i = 0; i < 8; ++i) z[i] = 0.f;
  return z;
}

// A fragment: 16x32 bf16: lanes 0-15 row M=lane, K=i+(lane>=16 ? 16:0)
__device__ __forceinline__ v16bf load_a_frag(const __bf16* Xs, int m0, int kk, int lane) {
  const __bf16* p = Xs + (m0 + (lane & 15)) * LDX + kk + ((lane >> 4) << 4);
  v16bf a;
#pragma unroll
  for (int i = 0; i < 8; ++i) { a[i] = p[i]; a[i + 8] = p[8 + i]; }
  return a;
}

// B fragment: 32x16 bf16; Bs stored N-major ([n][k]) -> 16 contiguous bf16 per lane.
__device__ __forceinline__ v16bf load_b_frag(const __bf16* Bs, int n0, int kk, int lane) {
  const __bf16* p = Bs + (n0 + (lane & 15)) * LDB + kk + ((lane >> 4) << 4);
  v16bf b;
#pragma unroll
  for (int i = 0; i < 16; ++i) b[i] = p[i];
  return b;
}

// Transpose-stage one 64k x 128n bf16 tile: 4x4 register-block transpose,
// coalesced uint2 global loads, ds_store_b64 LDS stores.
__device__ __forceinline__ void stage_b_tile(const __bf16* __restrict__ W, int ldW,
                                             int k0, int nc, __bf16* __restrict__ Bs,
                                             int tid, int Kdim) {
#pragma unroll
  for (int p = 0; p < 2; ++p) {
    int kg = (p * 8 + (tid >> 5)) * 4;
    int n4 = (tid & 31) * 4;
    const __bf16* gb = W + (size_t)(k0 + kg) * ldW + nc + n4;
    uint2 r0 = *reinterpret_cast<const uint2*>(gb);
    uint2 r1 = *reinterpret_cast<const uint2*>(gb + ldW);
    uint2 r2 = *reinterpret_cast<const uint2*>(gb + 2 * (size_t)ldW);
    uint2 r3 = *reinterpret_cast<const uint2*>(gb + 3 * (size_t)ldW);
    if (k0 + TILE_K < Kdim) __builtin_prefetch(gb + (size_t)TILE_K * ldW, 0, 1);
    const __bf16* b0 = reinterpret_cast<const __bf16*>(&r0);
    const __bf16* b1 = reinterpret_cast<const __bf16*>(&r1);
    const __bf16* b2 = reinterpret_cast<const __bf16*>(&r2);
    const __bf16* b3 = reinterpret_cast<const __bf16*>(&r3);
#pragma unroll
    for (int nj = 0; nj < 4; ++nj) {
      union { __bf16 h[4]; uint2 u; } c;
      c.h[0] = b0[nj]; c.h[1] = b1[nj]; c.h[2] = b2[nj]; c.h[3] = b3[nj];
      *reinterpret_cast<uint2*>(&Bs[(n4 + nj) * LDB + kg]) = c.u;
    }
  }
}

// ---------------- f32 -> bf16 conversion pass ----------------
__global__ __launch_bounds__(256)
void cvt_f32_to_bf16(const float* __restrict__ s, __bf16* __restrict__ d, int n) {
  int i = (blockIdx.x * blockDim.x + threadIdx.x) * 4;
  int stride = gridDim.x * blockDim.x * 4;
  for (; i < n; i += stride) {
    float4 v = *reinterpret_cast<const float4*>(s + i);
    __bf16 o[4] = { (__bf16)v.x, (__bf16)v.y, (__bf16)v.z, (__bf16)v.w };
    *reinterpret_cast<uint2*>(d + i) = *reinterpret_cast<const uint2*>(o);
  }
}

// ---------------- router: softmax over 8, top-4, expert gather lists ----------------
__global__ __launch_bounds__(256)
void router_kernel(const float* __restrict__ hs, const float* __restrict__ rw,
                   const float* __restrict__ rbias,
                   int* __restrict__ eCnt, int* __restrict__ eRows, float* __restrict__ eWt) {
  __shared__ float lg[32][NEXP];
  const int tid = threadIdx.x;
  const int tl = tid >> 3, e = tid & 7;
  const int token = blockIdx.x * 32 + tl;
  const float* x = hs + (size_t)token * DIM_H;
  float s = 0.f;
  for (int h = 0; h < DIM_H; h += 4) {
    float4 xv = *reinterpret_cast<const float4*>(x + h);
    s += xv.x * rw[(h + 0) * NEXP + e] + xv.y * rw[(h + 1) * NEXP + e] +
         xv.z * rw[(h + 2) * NEXP + e] + xv.w * rw[(h + 3) * NEXP + e];
  }
  lg[tl][e] = s;
  __syncthreads();
  if (tid < 32) {
    const int tok2 = blockIdx.x * 32 + tid;
    float l[NEXP], sc[NEXP];
    float mx = -3.4e38f;
#pragma unroll
    for (int j = 0; j < NEXP; ++j) { l[j] = lg[tid][j]; mx = fmaxf(mx, l[j]); }
    float den = 0.f;
#pragma unroll
    for (int j = 0; j < NEXP; ++j) { sc[j] = __expf(l[j] - mx); den += sc[j]; }
    float inv = 1.f / den;
#pragma unroll
    for (int j = 0; j < NEXP; ++j) sc[j] *= inv;
    const float bias = rbias[0];
    bool used[NEXP] = {};
    int idx[TOPK]; float wts[TOPK]; float wsum = 0.f;
#pragma unroll
    for (int k = 0; k < TOPK; ++k) {
      int best = 0; float bv = -3.4e38f;
#pragma unroll
      for (int j = 0; j < NEXP; ++j) {
        float c = sc[j] + bias;
        if (!used[j] && c > bv) { bv = c; best = j; }
      }
      used[best] = true; idx[k] = best; wts[k] = sc[best]; wsum += sc[best];
    }
    float norm = 1.f / (wsum + 1e-20f);   // NORM_TOPK; ROUTED_SCALING == 1
#pragma unroll
    for (int k = 0; k < TOPK; ++k) {
      float wk = wts[k] * norm;
      int p = atomicAdd(&eCnt[idx[k]], 1);
      eRows[idx[k] * MAXR + p] = tok2;
      eWt[idx[k] * MAXR + p] = wk;
    }
  }
}

// ---------------- fused gate/up GEMMs + SwiGLU, routing weight folded into act ----------------
__global__ __launch_bounds__(256)
void ffn_gateup_kernel(const __bf16* __restrict__ X, int ldX,
                       const __bf16* __restrict__ WgAll,
                       const __bf16* __restrict__ WuAll,
                       __bf16* __restrict__ act, int ldAct,
                       const int* __restrict__ rowList,   // null => identity rows
                       const float* __restrict__ wList,   // null => weight 1.0
                       const int* __restrict__ cntAll,    // null => totalRows
                       int Kdim, int Ndim, int totalRows, int tilesPer) {
  __shared__ __bf16 Xs[TILE_M * LDX];
  __shared__ __bf16 Bg[TILE_N * LDB];
  __shared__ __bf16 Bu[TILE_N * LDB];
  __shared__ int   sTok[TILE_M];
  __shared__ float sWt[TILE_M];

  const int e    = blockIdx.x / tilesPer;
  const int tile = blockIdx.x % tilesPer;
  const int row0 = tile * TILE_M;
  const int cnt  = cntAll ? cntAll[e] : totalRows;
  if (row0 >= cnt) return;
  const int rows = min(TILE_M, cnt - row0);

  const size_t wOff = (size_t)e * Kdim * Ndim;
  const __bf16* Wg = WgAll + wOff;
  const __bf16* Wu = WuAll + wOff;
  const int listBase = e * MAXR + row0;
  const size_t actBase = (size_t)listBase;

  const int tid = threadIdx.x, lane = tid & 31, w = tid >> 5;
  if (tid < TILE_M) {
    int tok = -1; float wt = 1.f;
    if (tid < rows) {
      tok = rowList ? rowList[listBase + tid] : (row0 + tid);
      if (wList) wt = wList[listBase + tid];
    }
    sTok[tid] = tok; sWt[tid] = wt;
  }
  // Pre-zero pad rows of Xs once: with async LDS loads, masked lanes never write them.
  if (rows < TILE_M) {
    for (int idx = tid; idx < (TILE_M - rows) * 8; idx += 256) {
      int rr = rows + (idx >> 3);
      uint4 z = {0u, 0u, 0u, 0u};
      *reinterpret_cast<uint4*>(&Xs[rr * LDX + (idx & 7) * 8]) = z;
    }
  }
  __syncthreads();

  // per-thread A-staging coordinates (fixed across k stages)
  const int rStage = tid >> 2, csStage = (tid & 3) * 16;
  const int tokStage = sTok[rStage];
  const __bf16* aRow = (tokStage >= 0) ? (X + (size_t)tokStage * ldX + csStage) : nullptr;
  __bf16* aDst = &Xs[rStage * LDX + csStage];

  const int m0 = (w & 3) * 16;
  const int nhalf = (w >> 2) * 64;

  for (int nc = 0; nc < Ndim; nc += TILE_N) {
    v8f accg[4], accu[4];
#pragma unroll
    for (int j = 0; j < 4; ++j) { accg[j] = zero8(); accu[j] = zero8(); }

    for (int k0 = 0; k0 < Kdim; k0 += TILE_K) {
      __syncthreads();
      // ---- stage A tile 64x64 bf16 (gathered rows) ----
      if (aRow) {
#ifdef HAVE_ASYNC_LDS
        __builtin_amdgcn_global_load_async_to_lds_b128(GLB_V4I(aRow + k0), LDS_V4I(aDst), 0, 0);
        __builtin_amdgcn_global_load_async_to_lds_b128(GLB_V4I(aRow + k0 + 8), LDS_V4I(aDst), 16, 0);
#else
        const uint4* g = reinterpret_cast<const uint4*>(aRow + k0);
        uint4 v0 = g[0], v1 = g[1];
        uint4* d = reinterpret_cast<uint4*>(aDst);
        d[0] = v0; d[1] = v1;
#endif
        __builtin_prefetch(aRow + k0 + TILE_K, 0, 1);
      }
      // ---- stage B tiles transposed (N-major) ----
      stage_b_tile(Wg, Ndim, k0, nc, Bg, tid, Kdim);
      stage_b_tile(Wu, Ndim, k0, nc, Bu, tid, Kdim);
#ifdef HAVE_ASYNC_LDS
      wait_async0();
#endif
      __syncthreads();
      // ---- WMMA ----
#pragma unroll
      for (int kk = 0; kk < TILE_K; kk += 32) {
        v16bf a = load_a_frag(Xs, m0, kk, lane);
#pragma unroll
        for (int j = 0; j < 4; ++j) {
          v16bf bg = load_b_frag(Bg, nhalf + j * 16, kk, lane);
          accg[j] = __builtin_amdgcn_wmma_f32_16x16x32_bf16(false, a, false, bg,
                                                            (short)0, accg[j], false, false);
          v16bf bu = load_b_frag(Bu, nhalf + j * 16, kk, lane);
          accu[j] = __builtin_amdgcn_wmma_f32_16x16x32_bf16(false, a, false, bu,
                                                            (short)0, accu[j], false, false);
        }
      }
    }
    // SwiGLU epilogue; fold routing weight (down GEMM is linear); store act as bf16
    const int hi = lane >> 4, col = lane & 15;
#pragma unroll
    for (int j = 0; j < 4; ++j) {
      int n = nc + nhalf + j * 16 + col;
#pragma unroll
      for (int r = 0; r < 8; ++r) {
        int M = m0 + r + hi * 8;
        float g = accg[j][r], u = accu[j][r];
        float a = (g / (1.f + __expf(-g))) * u * sWt[M];
        act[(actBase + M) * (size_t)ldAct + n] = (__bf16)a;
      }
    }
  }
}

// ---------------- down GEMM + scatter-accumulate into output ----------------
__global__ __launch_bounds__(256)
void ffn_down_kernel(const __bf16* __restrict__ act, int ldAct,
                     const __bf16* __restrict__ WdAll,
                     float* __restrict__ out,
                     const int* __restrict__ rowList,
                     const int* __restrict__ cntAll,
                     int Kdim, int Ndim, int totalRows, int tilesPer) {
  __shared__ __bf16 Xs[TILE_M * LDX];
  __shared__ __bf16 Bd[TILE_N * LDB];
  __shared__ int sTok[TILE_M];

  const int e    = blockIdx.x / tilesPer;
  const int tile = blockIdx.x % tilesPer;
  const int row0 = tile * TILE_M;
  const int cnt  = cntAll ? cntAll[e] : totalRows;
  if (row0 >= cnt) return;
  const int rows = min(TILE_M, cnt - row0);

  const __bf16* Wd = WdAll + (size_t)e * Kdim * Ndim;
  const int listBase = e * MAXR + row0;
  const size_t actBase = (size_t)listBase;

  const int tid = threadIdx.x, lane = tid & 31, w = tid >> 5;
  if (tid < TILE_M) {
    sTok[tid] = (tid < rows) ? (rowList ? rowList[listBase + tid] : (row0 + tid)) : -1;
  }
  __syncthreads();

  // act rows are dense per expert slot and zero-padded by gateup -> unconditional copy
  const int rStage = tid >> 2, csStage = (tid & 3) * 16;
  const __bf16* aRow = act + (actBase + rStage) * (size_t)ldAct + csStage;
  __bf16* aDst = &Xs[rStage * LDX + csStage];

  const int m0 = (w & 3) * 16;
  const int nhalf = (w >> 2) * 64;

  for (int nc = 0; nc < Ndim; nc += TILE_N) {
    v8f acc[4];
#pragma unroll
    for (int j = 0; j < 4; ++j) acc[j] = zero8();

    for (int k0 = 0; k0 < Kdim; k0 += TILE_K) {
      __syncthreads();
#ifdef HAVE_ASYNC_LDS
      __builtin_amdgcn_global_load_async_to_lds_b128(GLB_V4I(aRow + k0), LDS_V4I(aDst), 0, 0);
      __builtin_amdgcn_global_load_async_to_lds_b128(GLB_V4I(aRow + k0 + 8), LDS_V4I(aDst), 16, 0);
#else
      {
        const uint4* g = reinterpret_cast<const uint4*>(aRow + k0);
        uint4 v0 = g[0], v1 = g[1];
        uint4* d = reinterpret_cast<uint4*>(aDst);
        d[0] = v0; d[1] = v1;
      }
#endif
      stage_b_tile(Wd, Ndim, k0, nc, Bd, tid, Kdim);
#ifdef HAVE_ASYNC_LDS
      wait_async0();
#endif
      __syncthreads();
#pragma unroll
      for (int kk = 0; kk < TILE_K; kk += 32) {
        v16bf a = load_a_frag(Xs, m0, kk, lane);
#pragma unroll
        for (int j = 0; j < 4; ++j) {
          v16bf b = load_b_frag(Bd, nhalf + j * 16, kk, lane);
          acc[j] = __builtin_amdgcn_wmma_f32_16x16x32_bf16(false, a, false, b,
                                                           (short)0, acc[j], false, false);
        }
      }
    }
    const int hi = lane >> 4, col = lane & 15;
#pragma unroll
    for (int j = 0; j < 4; ++j) {
      int n = nc + nhalf + j * 16 + col;
#pragma unroll
      for (int r = 0; r < 8; ++r) {
        int M = m0 + r + hi * 8;
        if (M < rows) {
          int tok = sTok[M];
          unsafeAtomicAdd(&out[(size_t)tok * Ndim + n], acc[j][r]);
        }
      }
    }
  }
}

// ---------------- host-side orchestration ----------------
extern "C" void kernel_launch(void* const* d_in, const int* in_sizes, int n_in,
                              void* d_out, int out_size, void* d_ws, size_t ws_size,
                              hipStream_t stream) {
  (void)in_sizes; (void)n_in; (void)out_size; (void)ws_size;
  const float* hs = (const float*)d_in[0];
  const float* rw = (const float*)d_in[1];
  const float* rb = (const float*)d_in[2];
  const float* gw = (const float*)d_in[3];
  const float* uw = (const float*)d_in[4];
  const float* dw = (const float*)d_in[5];
  const float* sg = (const float*)d_in[6];
  const float* su = (const float*)d_in[7];
  const float* sd = (const float*)d_in[8];
  float* out = (float*)d_out;

  char* ws = (char*)d_ws;
  size_t off = 0;
  auto alloc = [&](size_t bytes) -> void* {
    void* p = ws + off;
    off += (bytes + 255) & ~(size_t)255;
    return p;
  };
  __bf16* hfb = (__bf16*)alloc((size_t)TOK_N * DIM_H * 2);
  __bf16* gwb = (__bf16*)alloc((size_t)NEXP * DIM_H * DIM_I * 2);
  __bf16* uwb = (__bf16*)alloc((size_t)NEXP * DIM_H * DIM_I * 2);
  __bf16* dwb = (__bf16*)alloc((size_t)NEXP * DIM_I * DIM_H * 2);
  __bf16* sgb = (__bf16*)alloc((size_t)DIM_H * DIM_IS * 2);
  __bf16* sub = (__bf16*)alloc((size_t)DIM_H * DIM_IS * 2);
  __bf16* sdb = (__bf16*)alloc((size_t)DIM_IS * DIM_H * 2);
  __bf16* actR = (__bf16*)alloc((size_t)NEXP * MAXR * DIM_I * 2);  // 16 MiB
  __bf16* actS = actR;  // shared-expert act (8 MiB) reuses routed act: stream-ordered, safe
  int*   eCnt  = (int*)alloc(NEXP * 4);
  int*   eRows = (int*)alloc((size_t)NEXP * MAXR * 4);
  float* eWt   = (float*)alloc((size_t)NEXP * MAXR * 4);

  (void)hipMemsetAsync(out, 0, (size_t)TOK_N * DIM_H * 4, stream);
  (void)hipMemsetAsync(eCnt, 0, NEXP * 4, stream);

  const int CVB = 2048, CVT = 256;
  cvt_f32_to_bf16<<<CVB, CVT, 0, stream>>>(hs, hfb, TOK_N * DIM_H);
  cvt_f32_to_bf16<<<CVB, CVT, 0, stream>>>(gw, gwb, NEXP * DIM_H * DIM_I);
  cvt_f32_to_bf16<<<CVB, CVT, 0, stream>>>(uw, uwb, NEXP * DIM_H * DIM_I);
  cvt_f32_to_bf16<<<CVB, CVT, 0, stream>>>(dw, dwb, NEXP * DIM_I * DIM_H);
  cvt_f32_to_bf16<<<CVB, CVT, 0, stream>>>(sg, sgb, DIM_H * DIM_IS);
  cvt_f32_to_bf16<<<CVB, CVT, 0, stream>>>(su, sub, DIM_H * DIM_IS);
  cvt_f32_to_bf16<<<CVB, CVT, 0, stream>>>(sd, sdb, DIM_IS * DIM_H);

  router_kernel<<<TOK_N / 32, 256, 0, stream>>>(hs, rw, rb, eCnt, eRows, eWt);

  const int tilesPer = MAXR / TILE_M;  // 32
  // routed experts
  ffn_gateup_kernel<<<NEXP * tilesPer, 256, 0, stream>>>(
      hfb, DIM_H, gwb, uwb, actR, DIM_I, eRows, eWt, eCnt, DIM_H, DIM_I, MAXR, tilesPer);
  ffn_down_kernel<<<NEXP * tilesPer, 256, 0, stream>>>(
      actR, DIM_I, dwb, out, eRows, eCnt, DIM_I, DIM_H, MAXR, tilesPer);
  // shared expert (identity rows, unit weights)
  ffn_gateup_kernel<<<TOK_N / TILE_M, 256, 0, stream>>>(
      hfb, DIM_H, sgb, sub, actS, DIM_IS, nullptr, nullptr, nullptr, DIM_H, DIM_IS, TOK_N, TOK_N / TILE_M);
  ffn_down_kernel<<<TOK_N / TILE_M, 256, 0, stream>>>(
      actS, DIM_IS, sdb, out, nullptr, nullptr, DIM_IS, DIM_H, TOK_N, TOK_N / TILE_M);
}
